// PositionalMambaModel_86337432584642
// MI455X (gfx1250) — compile-verified, matrix operand order checked by reading, not run on hardware
//
#include <hip/hip_runtime.h>
#include <math.h>

// ---------------------------------------------------------------------------
// CDNA5 (gfx1250) types for WMMA
// ---------------------------------------------------------------------------
typedef __attribute__((ext_vector_type(16))) __bf16        v16bf;
typedef __attribute__((ext_vector_type(8)))  float         v8f;
typedef __attribute__((ext_vector_type(4)))  unsigned int  v4u;
typedef __attribute__((ext_vector_type(2)))  unsigned int  v2u;

__device__ __forceinline__ unsigned short f2bf(float f) {
  // round-to-nearest-even f32 -> bf16
  unsigned int u = __float_as_uint(f);
  u += 0x7FFFu + ((u >> 16) & 1u);
  return (unsigned short)(u >> 16);
}
__device__ __forceinline__ unsigned int pack2bf(float a, float b) {
  return (unsigned int)f2bf(a) | ((unsigned int)f2bf(b) << 16);
}
__device__ __forceinline__ float sigmoidf_(float x) { return 1.f / (1.f + expf(-x)); }

// ---------------------------------------------------------------------------
// bf16-WMMA GEMM:  C[M,N] = A[M,K] * B[K,N] (+ Cadd)
// A, B f32 in global; converted to bf16 while staging into LDS.
// Block: 256 thr = 8 waves. Block tile 128x128, wave tile 32x64, K-step 64.
// A staged row-major As[row][k]; B staged TRANSPOSED BsT[col][k] so every
// fragment is two contiguous ds_load_b128 per ISA 7.12.2 lane layout.
// ---------------------------------------------------------------------------
#define ASTR 72   // LDS row stride (bf16 elems), 144 B (16B-aligned rows)
#define BSTR 72

__global__ __launch_bounds__(256) void gemm_bf16wmma(
    const float* __restrict__ A, int lda,
    const float* __restrict__ Bg, int ldb,
    float* __restrict__ C, int ldc,
    const float* __restrict__ Cadd,   // optional residual, may be null
    int M, int N, int K)
{
  __shared__ unsigned short As [128 * ASTR];   // 128 rows x 64 k
  __shared__ unsigned short BsT[128 * BSTR];   // 128 cols x 64 k

  const int t    = threadIdx.x;
  const int lane = t & 31;
  const int wv   = t >> 5;
  const int wm   = wv & 3;        // 0..3 -> 32-row stripe
  const int wn   = wv >> 2;       // 0..1 -> 64-col stripe
  const int m0   = blockIdx.y * 128;
  const int n0   = blockIdx.x * 128;

  const v8f vzero = {0.f, 0.f, 0.f, 0.f, 0.f, 0.f, 0.f, 0.f};
  v8f acc[2][4];
  for (int i = 0; i < 2; ++i)
    for (int j = 0; j < 4; ++j)
      acc[i][j] = vzero;

  const int lr = lane & 15;          // row/col within 16-tile
  const int kb = (lane >> 4) * 8;    // lanes 16..31 handle K+8 block

  for (int k0 = 0; k0 < K; k0 += 64) {
    // ================= stage A tile (128 x 64) =================
    if (m0 + 128 <= M && k0 + 64 <= K) {
      // fast path: float4 global loads, 8B packed LDS stores
      #pragma unroll
      for (int i = 0; i < 8; ++i) {
        int idx = i * 256 + t;             // 2048 float4 slots
        int r = idx >> 4, c4 = idx & 15;
        float4 v = *(const float4*)&A[(size_t)(m0 + r) * lda + k0 + c4 * 4];
        v2u p; p.x = pack2bf(v.x, v.y); p.y = pack2bf(v.z, v.w);
        *(v2u*)&As[r * ASTR + c4 * 4] = p;
      }
    } else {
      #pragma unroll
      for (int i = 0; i < 32; ++i) {
        int idx = i * 256 + t;             // 8192 elems
        int r = idx >> 6, c = idx & 63;
        int gm = m0 + r, gk = k0 + c;
        unsigned short v = 0;
        if (gm < M && gk < K) v = f2bf(A[(size_t)gm * lda + gk]);
        As[r * ASTR + c] = v;
      }
    }
    // ================= stage B tile (64 x 128), transposed =================
    if (n0 + 128 <= N && k0 + 64 <= K) {
      #pragma unroll
      for (int i = 0; i < 8; ++i) {
        int idx = i * 256 + t;             // 2048 (col,k4) slots
        int col = idx & 127, k4 = idx >> 7;
        const float* bp = &Bg[(size_t)(k0 + k4 * 4) * ldb + n0 + col];
        float b0 = bp[0];
        float b1 = bp[(size_t)ldb];
        float b2 = bp[(size_t)2 * ldb];
        float b3 = bp[(size_t)3 * ldb];
        v2u p; p.x = pack2bf(b0, b1); p.y = pack2bf(b2, b3);
        *(v2u*)&BsT[col * BSTR + k4 * 4] = p;
      }
    } else {
      #pragma unroll
      for (int i = 0; i < 32; ++i) {
        int idx = i * 256 + t;             // 8192 elems
        int col = idx & 127, kk = idx >> 7;
        int gk = k0 + kk, gn = n0 + col;
        unsigned short v = 0;
        if (gk < K && gn < N) v = f2bf(Bg[(size_t)gk * ldb + gn]);
        BsT[col * BSTR + kk] = v;
      }
    }
    __syncthreads();

    if (k0 + 64 < K) {  // cache hint for next tiles (global_prefetch_b8)
      __builtin_prefetch(&A [(size_t)(m0 + (t >> 1)) * lda + k0 + 64], 0, 1);
      __builtin_prefetch(&Bg[(size_t)(k0 + 64 + (t >> 4)) * ldb + n0 + (t & 15) * 8], 0, 1);
    }

    // ================= 2 WMMA sub-steps of K=32 =================
    #pragma unroll
    for (int s = 0; s < 2; ++s) {
      const int ks = s * 32;
      union Frag { v16bf v; v4u q[2]; } fa[2], fb[4];
      #pragma unroll
      for (int i = 0; i < 2; ++i) {
        int row = wm * 32 + i * 16 + lr;
        fa[i].q[0] = *(const v4u*)&As[row * ASTR + ks + kb];
        fa[i].q[1] = *(const v4u*)&As[row * ASTR + ks + 16 + kb];
      }
      #pragma unroll
      for (int j = 0; j < 4; ++j) {
        int col = wn * 64 + j * 16 + lr;
        fb[j].q[0] = *(const v4u*)&BsT[col * BSTR + ks + kb];
        fb[j].q[1] = *(const v4u*)&BsT[col * BSTR + ks + 16 + kb];
      }
      #pragma unroll
      for (int i = 0; i < 2; ++i)
        #pragma unroll
        for (int j = 0; j < 4; ++j)
          acc[i][j] = __builtin_amdgcn_wmma_f32_16x16x32_bf16(
              false, fa[i].v, false, fb[j].v, (short)0, acc[i][j], false, false);
    }
    __syncthreads();
  }

  // ---- store (C/D layout: VGPR r -> M=r, +8 for lanes 16..31) ----
  const int mo = (lane >> 4) * 8;
  for (int i = 0; i < 2; ++i)
    for (int j = 0; j < 4; ++j)
      #pragma unroll
      for (int r = 0; r < 8; ++r) {
        int m = m0 + wm * 32 + i * 16 + mo + r;
        int n = n0 + wn * 64 + j * 16 + lr;
        if (m < M && n < N) {
          float v = acc[i][j][r];
          if (Cadd) v += Cadd[(size_t)m * ldc + n];
          C[(size_t)m * ldc + n] = v;
        }
      }
}

// ---------------------------------------------------------------------------
// Embedding gather + positional-feature projection (f=2) broadcast over 3.
// ---------------------------------------------------------------------------
__global__ void embed_kernel(const int* __restrict__ x, const float* __restrict__ features,
                             const float* __restrict__ emb, const float* __restrict__ fw,
                             const float* __restrict__ fb, float* __restrict__ h,
                             int B, int L, int D)
{
  int i = blockIdx.x * blockDim.x + threadIdx.x;
  int total = B * L * D;
  if (i >= total) return;
  int d = i % D;
  int bl = i / D;
  int l = bl % L, b = bl / L;
  int tok = x[b * L + l];
  const float* f = features + ((size_t)b * (L / 3) + (l / 3)) * 2;
  float pf = f[0] * fw[d] + f[1] * fw[D + d] + fb[d];
  h[i] = emb[(size_t)tok * D + d] + pf;
}

// ---------------------------------------------------------------------------
// RMSNorm: one block (256 thr) per row of D elements.
// ---------------------------------------------------------------------------
__global__ __launch_bounds__(256) void rmsnorm_kernel(const float* __restrict__ x,
                                                      const float* __restrict__ w,
                                                      float* __restrict__ out, int D)
{
  __shared__ float red[256];
  int row = blockIdx.x;
  const float* xr = x + (size_t)row * D;
  float s = 0.f;
  for (int d = threadIdx.x; d < D; d += 256) { float v = xr[d]; s += v * v; }
  red[threadIdx.x] = s;
  __syncthreads();
  for (int o = 128; o > 0; o >>= 1) {
    if (threadIdx.x < o) red[threadIdx.x] += red[threadIdx.x + o];
    __syncthreads();
  }
  float inv = rsqrtf(red[0] / (float)D + 1e-5f);
  for (int d = threadIdx.x; d < D; d += 256)
    out[(size_t)row * D + d] = xr[d] * inv * w[d];
}

// ---------------------------------------------------------------------------
// Depthwise causal conv (k=4) + bias + SiLU over the u half of xr.
// ---------------------------------------------------------------------------
__global__ void conv_silu_kernel(const float* __restrict__ xr, const float* __restrict__ cw,
                                 const float* __restrict__ cb, float* __restrict__ u,
                                 int B, int L, int DI)
{
  int i = blockIdx.x * blockDim.x + threadIdx.x;
  int total = B * L * DI;
  if (i >= total) return;
  int d = i % DI;
  int bl = i / DI;
  int l = bl % L, b = bl / L;
  float s = cb[d];
  #pragma unroll
  for (int j = 0; j < 4; ++j) {
    int ls = l - 3 + j;
    if (ls >= 0) s += cw[d * 4 + j] * xr[((size_t)(b * L + ls)) * 3072 + d];
  }
  u[i] = s * sigmoidf_(s);
}

// ---------------------------------------------------------------------------
// Selective-scan: 4 lanes per (b,d) channel, 4 states each (16 total).
// y reduced across the 4 lanes with a shfl_xor butterfly.  z may alias dr.
// ---------------------------------------------------------------------------
__global__ __launch_bounds__(256) void scan_kernel(
    const float* __restrict__ dr, const float* __restrict__ dtb,
    const float* __restrict__ u, const float* __restrict__ xdbl,
    const float* __restrict__ A_log, const float* __restrict__ Dp,
    const float* __restrict__ xr, float* __restrict__ z,
    int B, int L, int DI)
{
  int t = blockIdx.x * blockDim.x + threadIdx.x;
  if (t >= B * DI * 4) return;
  const int sub = t & 3;          // which 4-state slice
  const int g   = t >> 2;
  const int b = g / DI, d = g % DI;

  float Ai[4], hst[4];
  #pragma unroll
  for (int q = 0; q < 4; ++q) {
    int n = sub * 4 + q;
    Ai[q]  = -expf(A_log[(size_t)d * 16 + n]);
    hst[q] = 0.f;
  }
  const float db = dtb[d];
  const float dp = Dp[d];

  for (int l = 0; l < L; ++l) {
    size_t base = (size_t)(b * L + l);
    float xv = dr[base * DI + d] + db;
    float delta = (xv > 20.f) ? xv : log1pf(expf(xv));
    float ut = u[base * DI + d];
    const float* xd = xdbl + base * 80;
    float y = 0.f;
    #pragma unroll
    for (int q = 0; q < 4; ++q) {
      int n = sub * 4 + q;
      float dA = expf(delta * Ai[q]);
      hst[q] = dA * hst[q] + delta * xd[48 + n] * ut;
      y += hst[q] * xd[64 + n];
    }
    // sum partial y across the 4 lanes of this channel
    y += __shfl_xor(y, 1);
    y += __shfl_xor(y, 2);
    if (sub == 0) {
      float res = xr[base * 3072 + DI + d];
      z[base * DI + d] = (y + ut * dp) * (res * sigmoidf_(res));
    }
  }
}

// ---------------------------------------------------------------------------
// Positional vocab mask (in-place on logits).
// ---------------------------------------------------------------------------
__global__ void mask_kernel(float* __restrict__ out, int B, int L, int V)
{
  int i = blockIdx.x * blockDim.x + threadIdx.x;
  int total = B * L * V;
  if (i >= total) return;
  int v = i % V;
  int l = (i / V) % L;
  int p = l % 3;
  bool keep = (p == 0) ? (v >= 1 && v < 10)
            : (p == 1) ? (v >= 100 && v < 200)
                       : (v >= 200 && v < 300);
  if (!keep) out[i] = -1000000000.0f;
}

// ---------------------------------------------------------------------------
// Host launcher
// ---------------------------------------------------------------------------
extern "C" void kernel_launch(void* const* d_in, const int* in_sizes, int n_in,
                              void* d_out, int out_size, void* d_ws, size_t ws_size,
                              hipStream_t stream)
{
  (void)in_sizes; (void)n_in; (void)out_size; (void)ws_size;

  constexpr int B = 8, L = 1536, D = 768, V = 512;
  constexpr int NL = 4, DI = 1536, DTR = 48, DS_N = 16, DCONV = 4;
  constexpr int M = B * L;                 // 12288 rows

  const int*   x         = (const int*)  d_in[0];
  const float* features  = (const float*)d_in[1];
  const float* emb       = (const float*)d_in[2];
  const float* fproj_w   = (const float*)d_in[3];
  const float* fproj_b   = (const float*)d_in[4];
  const float* norm_w    = (const float*)d_in[5];
  const float* in_w      = (const float*)d_in[6];
  const float* conv_w    = (const float*)d_in[7];
  const float* conv_b    = (const float*)d_in[8];
  const float* xp_w      = (const float*)d_in[9];
  const float* dt_w      = (const float*)d_in[10];
  const float* dt_b      = (const float*)d_in[11];
  const float* A_log     = (const float*)d_in[12];
  const float* Dp        = (const float*)d_in[13];
  const float* out_w     = (const float*)d_in[14];
  const float* norm_f_w  = (const float*)d_in[15];
  const float* lm_w      = (const float*)d_in[16];
  float*       out       = (float*)d_out;

  // workspace layout (f32)
  float* ws   = (float*)d_ws;
  float* h    = ws;                              // M*768
  float* xn   = h    + (size_t)M * D;            // M*768
  float* xr   = xn   + (size_t)M * D;            // M*3072
  float* u    = xr   + (size_t)M * 2 * DI;       // M*1536
  float* xdbl = u    + (size_t)M * DI;           // M*80
  float* dr   = xdbl + (size_t)M * 80;           // M*1536 (delta_raw, reused as z)

  const dim3 blk(256);

  // h = emb[x] + repeat(features @ fproj_w + fproj_b, 3)
  embed_kernel<<<dim3((M * D + 255) / 256), blk, 0, stream>>>(
      x, features, emb, fproj_w, fproj_b, h, B, L, D);

  const dim3 gIn ((3072 + 127) / 128, (M + 127) / 128);
  const dim3 gXp ((80   + 127) / 128, (M + 127) / 128);
  const dim3 gDt ((DI   + 127) / 128, (M + 127) / 128);
  const dim3 gOut((D    + 127) / 128, (M + 127) / 128);
  const dim3 gLm ((V    + 127) / 128, (M + 127) / 128);

  for (int i = 0; i < NL; ++i) {
    // xn = rmsnorm(h, norm_w[i])
    rmsnorm_kernel<<<M, blk, 0, stream>>>(h, norm_w + (size_t)i * D, xn, D);

    // xr = xn @ in_proj_w[i]   (M x 3072)
    gemm_bf16wmma<<<gIn, blk, 0, stream>>>(
        xn, D, in_w + (size_t)i * D * 2 * DI, 2 * DI,
        xr, 2 * DI, nullptr, M, 2 * DI, D);

    // u = silu(causal_conv(xr[:, :DI]) + conv_b)
    conv_silu_kernel<<<dim3((M * DI + 255) / 256), blk, 0, stream>>>(
        xr, conv_w + (size_t)i * DI * DCONV, conv_b + (size_t)i * DI, u, B, L, DI);

    // xdbl = u @ x_proj_w[i]   (M x 80)
    gemm_bf16wmma<<<gXp, blk, 0, stream>>>(
        u, DI, xp_w + (size_t)i * DI * 80, 80,
        xdbl, 80, nullptr, M, 80, DI);

    // dr = xdbl[:, :48] @ dt_proj_w[i]   (M x 1536)
    gemm_bf16wmma<<<gDt, blk, 0, stream>>>(
        xdbl, 80, dt_w + (size_t)i * DTR * DI, DI,
        dr, DI, nullptr, M, DI, DTR);

    // z = scan(...)  (in-place over dr)
    scan_kernel<<<dim3((B * DI * 4 + 255) / 256), blk, 0, stream>>>(
        dr, dt_b + (size_t)i * DI, u, xdbl,
        A_log + (size_t)i * DI * DS_N, Dp + (size_t)i * DI,
        xr, dr, B, L, DI);

    // h = z @ out_proj_w[i] + h   (fused residual)
    gemm_bf16wmma<<<gOut, blk, 0, stream>>>(
        dr, DI, out_w + (size_t)i * DI * D, D,
        h, D, h, M, D, DI);
  }

  // final norm + lm head + positional mask
  rmsnorm_kernel<<<M, blk, 0, stream>>>(h, norm_f_w, xn, D);
  gemm_bf16wmma<<<gLm, blk, 0, stream>>>(
      xn, D, lm_w, V, out, V, nullptr, M, V, D);
  mask_kernel<<<dim3((M * V + 255) / 256), blk, 0, stream>>>(out, B, L, V);
}
